// RIMCell_4329327034454
// MI455X (gfx1250) — matrix operand order repaired and checked by reading, not compiled, not run on hardware
//
#include <hip/hip_runtime.h>
#include <hip/hip_bf16.h>
#include <stdint.h>
#include <math.h>

// ---------------- problem constants (from reference) ----------------
constexpr int Bn   = 4096;
constexpr int Un   = 6;
constexpr int INn  = 256;
constexpr int Hn   = 600;
constexpr int IKS  = 64;
constexpr int IVS  = 400;
constexpr int NCH  = 4;
constexpr int CVS  = Hn;          // 600
constexpr int TOPK = 4;

// padded dims (K -> mult of 32, used as GEMM A row length)
constexpr int Hp   = 608;         // 600 -> 608
constexpr int IVSp = 416;         // 400 -> 416

typedef unsigned long long ull;
typedef unsigned short     u16;

// ---------------- bf16 WMMA types ----------------
typedef __attribute__((ext_vector_type(16))) __bf16 v16bf;
typedef __attribute__((ext_vector_type(8)))  float  v8f;

union Frag {
    uint4  u[2];
    v16bf  v;
};

__device__ __forceinline__ unsigned f2bf(float f) {
    unsigned u = __float_as_uint(f);
    return (u + 0x7FFFu + ((u >> 16) & 1u)) >> 16;   // RNE truncation
}

__device__ __forceinline__ float wred32(float v) {
#pragma unroll
    for (int off = 16; off > 0; off >>= 1)
        v += __shfl_xor(v, off, 32);
    return v;
}

__device__ __forceinline__ void async_cp16(unsigned ldsoff, const u16* g) {
    asm volatile("global_load_async_to_lds_b128 %0, %1, off"
                 :: "v"(ldsoff), "v"(g) : "memory");
}

// =====================================================================
// Activation conversion: [R x C] fp32 -> [R x Cp] bf16, zero-padded cols
// =====================================================================
__global__ void __launch_bounds__(256)
cvtA_kernel(const float* __restrict__ in, u16* __restrict__ outp,
            int cp4 /* Cp/4 */, int C, int total4)
{
    int i = blockIdx.x * blockDim.x + threadIdx.x;
    if (i >= total4) return;
    int row = i / cp4;
    int c4  = (i - row * cp4) * 4;
    uint2 d = {0u, 0u};
    if (c4 < C) {        // C multiple of 4
        float4 f = *(const float4*)&in[(size_t)row * C + c4];
        d.x = f2bf(f.x) | (f2bf(f.y) << 16);
        d.y = f2bf(f.z) | (f2bf(f.w) << 16);
    }
    *(uint2*)&outp[(size_t)row * (cp4 * 4) + c4] = d;
}

// =====================================================================
// Weight conversion: (U,K,N) fp32 -> K-tile-transposed bf16
//   Wt[u][kt][n][k2]  (kt = K/32 tiles, n padded to Np, k2 = 0..31)
// One thread per (u,kt,n): strided reads, 4x16B contiguous writes.
// =====================================================================
__global__ void __launch_bounds__(256)
cvtW_kernel(const float* __restrict__ in, u16* __restrict__ outp,
            int K, int N, int Kp32, int Np, int total)
{
    int tid = blockIdx.x * blockDim.x + threadIdx.x;
    if (tid >= total) return;
    int n   = tid % Np;
    int rem = tid / Np;
    int kt  = rem % Kp32;
    int u   = rem / Kp32;

    const bool nok = (n < N);
    const float* src = in + ((size_t)u * K + kt * 32) * N + n;
    unsigned wds[16];
#pragma unroll
    for (int w = 0; w < 16; ++w) {
        float fa = 0.f, fb = 0.f;
        if (nok) {
            if (kt * 32 + 2 * w     < K) fa = src[(size_t)(2 * w) * N];
            if (kt * 32 + 2 * w + 1 < K) fb = src[(size_t)(2 * w + 1) * N];
        }
        wds[w] = f2bf(fa) | (f2bf(fb) << 16);
    }
    u16* dst = outp + (((size_t)u * Kp32 + kt) * Np + n) * 32;
#pragma unroll
    for (int j = 0; j < 4; ++j)
        ((uint4*)dst)[j] = make_uint4(wds[4 * j], wds[4 * j + 1], wds[4 * j + 2], wds[4 * j + 3]);
}

// =====================================================================
// Batched (per-unit) bf16 GEMM:  C[u] = A[u] (M x Kp) * W[u] (Kp x N) [+bias]
//   A: bf16, K-contiguous rows, Kp multiple of 32 (zero padded)
//   Wt: bf16, K-tile-transposed [u][kt][Np][32] (zero padded)
// Double-buffered LDS, all staging via global_load_async_to_lds_b128.
// Block: 256 thr (8 waves); block tile 128x64; wave tile 16x64 (4 WMMA).
// =====================================================================
__global__ void __launch_bounds__(256)
gemm_bf16_wmma(const u16* __restrict__ A, ull a_u_off, int a_row_stride,
               const u16* __restrict__ Wt, ull w_u_off, int w_np,
               const float* __restrict__ bias,
               float* __restrict__ C, ull c_u_off, int c_row_stride,
               int N, int Kp)
{
    __shared__ __align__(16) u16 Alds[2][128][40];   // 2 x 128 rows x 32 K (pad->40)
    __shared__ __align__(16) u16 Blds[2][64][40];    // 2 x 64 cols  x 32 K (pad->40)

    const int u    = blockIdx.z;
    const int m0   = blockIdx.x * 128;
    const int n0   = blockIdx.y * 64;
    const int t    = threadIdx.x;
    const int wid  = t >> 5;
    const int lane = t & 31;
    const int mlo  = lane & 15;
    const int kh   = lane >> 4;

    const u16* a_base = A  + (ull)u * a_u_off;
    const u16* w_base = Wt + (ull)u * w_u_off;
    float*     c_base = C  + (ull)u * c_u_off;

    // per-thread 16B-chunk coordinates (A: 2 chunks, B: 1 chunk)
    const int rA = t >> 2, cA = (t & 3) * 8;         // A rows 0..63 (+64), cols 0..24
    const int nB = t >> 2, cB = (t & 3) * 8;         // B cols 0..63,  k 0..24

    const int nt = Kp >> 5;

    auto issue_tile = [&](int p, int kt) {
        const size_t kof = (size_t)kt * 32;
        async_cp16((unsigned)(uintptr_t)&Alds[p][rA][cA],
                   a_base + (size_t)(m0 + rA) * a_row_stride + kof + cA);
        async_cp16((unsigned)(uintptr_t)&Alds[p][64 + rA][cA],
                   a_base + (size_t)(m0 + 64 + rA) * a_row_stride + kof + cA);
        async_cp16((unsigned)(uintptr_t)&Blds[p][nB][cB],
                   w_base + ((size_t)kt * w_np + n0 + nB) * 32 + cB);
    };

    v8f acc[4] = {};

    issue_tile(0, 0);
    for (int kt = 0; kt < nt; ++kt) {
        const int p = kt & 1;
        asm volatile("s_wait_asynccnt 0x0" ::: "memory");
        __syncthreads();                    // tile p ready; prior reads of p^1 done
        if (kt + 1 < nt) issue_tile(p ^ 1, kt + 1);

        // ---- A fragment (ISA 16-bit A 16x32 per-lane layout) ----
        const int arow = wid * 16 + mlo;
        Frag fa;
        fa.u[0] = *(const uint4*)&Alds[p][arow][kh * 8];
        fa.u[1] = *(const uint4*)&Alds[p][arow][16 + kh * 8];

#pragma unroll
        for (int c = 0; c < 4; ++c) {
            const int bn = c * 16 + mlo;
            Frag fb;
            fb.u[0] = *(const uint4*)&Blds[p][bn][kh * 8];
            fb.u[1] = *(const uint4*)&Blds[p][bn][16 + kh * 8];
            acc[c] = __builtin_amdgcn_wmma_f32_16x16x32_bf16(
                false, fa.v, false, fb.v, (short)0, acc[c], false, false);
        }
    }

    // ---- epilogue: vgpr r -> row r + 8*kh, lane%16 -> col ----
#pragma unroll
    for (int c = 0; c < 4; ++c) {
        int col = n0 + c * 16 + mlo;
        if (col < N) {
            float bv = bias ? bias[col] : 0.f;
#pragma unroll
            for (int r = 0; r < 8; ++r) {
                int row = m0 + wid * 16 + kh * 8 + r;
                c_base[(size_t)row * c_row_stride + col] = acc[c][r] + bv;
            }
        }
    }
}

// =====================================================================
// scores -> top-4 mask -> softmax(2) -> inputs (bf16, padded IVSp)
// =====================================================================
__global__ void __launch_bounds__(256)
scores_kernel(const float* __restrict__ qry,   // (B,U,64)
              const float* __restrict__ key,   // (B,64)
              const float* __restrict__ bk,    // (64)
              const float* __restrict__ val,   // (B,400)
              const float* __restrict__ bv,    // (400)
              float* __restrict__ maskp,       // (B,U)
              u16*   __restrict__ inputsb)     // (B,U,IVSp) bf16
{
    const int lane = threadIdx.x & 31;
    const int b = blockIdx.x * 8 + (threadIdx.x >> 5);

    float s0[Un], s1[Un];
    const float k0 = key[(size_t)b * 64 + lane];
    const float k1 = key[(size_t)b * 64 + lane + 32];
    const float n0 = bk[lane];
    const float n1 = bk[lane + 32];
#pragma unroll
    for (int uu = 0; uu < Un; ++uu) {
        const float* q = qry + ((size_t)b * Un + uu) * 64;
        float q0 = q[lane], q1 = q[lane + 32];
        s0[uu] = wred32(q0 * k0 + q1 * k1) * 0.125f;   // / sqrt(64)
        s1[uu] = wred32(q0 * n0 + q1 * n1) * 0.125f;
    }
    float m[Un];
#pragma unroll
    for (int uu = 0; uu < Un; ++uu) {
        int rank = 0;
#pragma unroll
        for (int u2 = 0; u2 < Un; ++u2)
            rank += (s0[u2] > s0[uu]) || (s0[u2] == s0[uu] && u2 < uu);
        m[uu] = (rank < TOPK) ? 1.f : 0.f;
    }
    if (lane == 0) {
#pragma unroll
        for (int uu = 0; uu < Un; ++uu)
            maskp[(size_t)b * Un + uu] = m[uu];
    }
#pragma unroll
    for (int uu = 0; uu < Un; ++uu) {
        float mx = fmaxf(s0[uu], s1[uu]);
        float e0 = __expf(s0[uu] - mx), e1 = __expf(s1[uu] - mx);
        float inv = 1.f / (e0 + e1);
        float p0 = e0 * inv * m[uu];
        float p1 = e1 * inv * m[uu];
        u16* dst = inputsb + ((size_t)b * Un + uu) * IVSp;
        const float* vb = val + (size_t)b * IVS;
        for (int c = lane; c < IVSp; c += 32) {
            u16 o = 0;
            if (c < IVS) o = (u16)f2bf(p0 * vb[c] + p1 * bv[c]);
            dst[c] = o;
        }
    }
}

// =====================================================================
// GRU gate: h_new fp32 -> d_out, padded bf16 copy -> hsnewb (B,U,Hp)
// =====================================================================
__global__ void __launch_bounds__(256)
gru_kernel(const float* __restrict__ gx,   // (B,U,1800)
           const float* __restrict__ gh,   // (B,U,1800)
           const float* __restrict__ hs,   // (B,U,600)
           float* __restrict__ hnew,       // (B,U,600) -> d_out
           u16*   __restrict__ hnewb)      // (B,U,Hp) bf16
{
    int idx = blockIdx.x * blockDim.x + threadIdx.x;
    if (idx >= Bn * Un * Hp) return;
    int bu = idx / Hp, c = idx % Hp;
    if (c < Hn) {
        size_t g = (size_t)bu * 1800 + c;
        float ir = gx[g], ii = gx[g + 600], in_ = gx[g + 1200];
        float hr = gh[g], hi = gh[g + 600], hn_ = gh[g + 1200];
        float rg = 1.f / (1.f + __expf(-(ir + hr)));
        float ig = 1.f / (1.f + __expf(-(ii + hi)));
        float ng = tanhf(in_ + rg * hn_);
        float h  = hs[(size_t)bu * Hn + c];
        float o  = ng + ig * (h - ng);
        hnew[(size_t)bu * Hn + c] = o;
        hnewb[idx] = (u16)f2bf(o);
    } else {
        hnewb[idx] = 0;
    }
}

// =====================================================================
// 4-head 6x6 attention + ctx (bf16)  [one wave per b]
// =====================================================================
__global__ void __launch_bounds__(256)
attn_kernel(const float* __restrict__ q,     // (B,U,128)
            const float* __restrict__ kk,    // (B,U,128)
            const float* __restrict__ v,     // (B,U,2400)
            const float* __restrict__ maskp, // (B,U)
            u16*   __restrict__ ctxb)        // (B,U,2400) bf16
{
    __shared__ float attl[8][NCH][Un][Un];
    const int w    = threadIdx.x >> 5;
    const int lane = threadIdx.x & 31;
    const int b    = blockIdx.x * 8 + w;

    const float scale = 0.17677669529663687f;   // 1/sqrt(32)
    float maskv = (lane < Un) ? maskp[(size_t)b * Un + lane] : 0.f;

#pragma unroll
    for (int h = 0; h < NCH; ++h) {
        float qv[Un], kv[Un];
#pragma unroll
        for (int uu = 0; uu < Un; ++uu) {
            qv[uu] = q [((size_t)b * Un + uu) * 128 + h * 32 + lane];
            kv[uu] = kk[((size_t)b * Un + uu) * 128 + h * 32 + lane];
        }
        float srow[Un];
#pragma unroll
        for (int uu = 0; uu < Un; ++uu) {
#pragma unroll
            for (int nn = 0; nn < Un; ++nn) {
                float r = wred32(qv[uu] * kv[nn]) * scale;
                if (lane == uu) srow[nn] = r;
            }
        }
        if (lane < Un) {
            float mx = srow[0];
#pragma unroll
            for (int nn = 1; nn < Un; ++nn) mx = fmaxf(mx, srow[nn]);
            float den = 0.f, e[Un];
#pragma unroll
            for (int nn = 0; nn < Un; ++nn) { e[nn] = __expf(srow[nn] - mx); den += e[nn]; }
            float inv = maskv / den;
#pragma unroll
            for (int nn = 0; nn < Un; ++nn)
                attl[w][h][lane][nn] = e[nn] * inv;
        }
    }
    __syncthreads();

    for (int c = lane; c < NCH * CVS; c += 32) {
        int h = c / CVS;
        float vv[Un];
#pragma unroll
        for (int nn = 0; nn < Un; ++nn)
            vv[nn] = v[((size_t)b * Un + nn) * (NCH * CVS) + c];
#pragma unroll
        for (int uu = 0; uu < Un; ++uu) {
            float a = 0.f;
#pragma unroll
            for (int nn = 0; nn < Un; ++nn)
                a = fmaf(attl[w][h][uu][nn], vv[nn], a);
            ctxb[((size_t)b * Un + uu) * (NCH * CVS) + c] = (u16)f2bf(a);
        }
    }
}

// =====================================================================
// final blend  out = m*(ctxW + h_new) + (1-m)*h_old
// =====================================================================
__global__ void __launch_bounds__(256)
blend_kernel(const float* __restrict__ ctxw,  // (B,U,600)
             const float* __restrict__ hsold, // (B,U,600)
             const float* __restrict__ maskp, // (B,U)
             float* __restrict__ out)         // (B,U,600), currently h_new
{
    int idx = blockIdx.x * blockDim.x + threadIdx.x;
    if (idx >= Bn * Un * Hn) return;
    float m = maskp[idx / Hn];
    float hb = out[idx];
    out[idx] = m * (ctxw[idx] + hb) + (1.f - m) * hsold[idx];
}

// =====================================================================
// host side
// =====================================================================
static inline int padK(int k) { return (k + 31) & ~31; }
static inline int padN(int n) { return (n + 63) & ~63; }

static void gemm(hipStream_t s,
                 const u16* A, ull au, int ars,
                 const u16* Wt, ull wu, int wnp,
                 const float* bias,
                 float* C, ull cu, int crs,
                 int M, int N, int Kp, int Uz)
{
    dim3 grid(M / 128, (N + 63) / 64, Uz);
    gemm_bf16_wmma<<<grid, dim3(256), 0, s>>>(A, au, ars, Wt, wu, wnp, bias, C, cu, crs, N, Kp);
}

static void cvtW(hipStream_t s, const float* in, u16* outp, int Uc, int K, int N)
{
    int Kp32 = padK(K) / 32, Np = padN(N);
    int total = Uc * Kp32 * Np;
    cvtW_kernel<<<(total + 255) / 256, 256, 0, s>>>(in, outp, K, N, Kp32, Np, total);
}

static void cvtA(hipStream_t s, const float* in, u16* outp, int R, int C, int Cp)
{
    int cp4 = Cp / 4;
    int total4 = R * cp4;
    cvtA_kernel<<<(total4 + 255) / 256, 256, 0, s>>>(in, outp, cp4, C, total4);
}

extern "C" void kernel_launch(void* const* d_in, const int* in_sizes, int n_in,
                              void* d_out, int out_size, void* d_ws, size_t ws_size,
                              hipStream_t stream)
{
    const float* x     = (const float*)d_in[0];   // (B,1,256)
    const float* hs    = (const float*)d_in[1];   // (B,U,600)
    const float* Wk    = (const float*)d_in[2];   // (256,64)
    const float* bk    = (const float*)d_in[3];   // (64)
    const float* Wv    = (const float*)d_in[4];   // (256,400)
    const float* bv    = (const float*)d_in[5];   // (400)
    const float* Wq    = (const float*)d_in[6];   // (U,600,64)
    const float* x2h   = (const float*)d_in[7];   // (U,400,1800)
    const float* h2h   = (const float*)d_in[8];   // (U,600,1800)
    const float* Wq_   = (const float*)d_in[9];   // (U,600,128)
    const float* Wk_   = (const float*)d_in[10];  // (U,600,128)
    const float* Wv_   = (const float*)d_in[11];  // (U,600,2400)
    const float* Wout  = (const float*)d_in[12];  // (U,2400,600)
    float* out = (float*)d_out;                   // (B,U,600)

    // ---------- workspace: fp32 region then bf16 region ----------
    float* wsf = (float*)d_ws;
    size_t off = 0;
    float* key    = wsf + off; off += (size_t)Bn * 64;
    float* val    = wsf + off; off += (size_t)Bn * IVS;
    float* qry    = wsf + off; off += (size_t)Bn * Un * IKS;
    float* maskp  = wsf + off; off += (size_t)Bn * Un;
    float* qbuf   = wsf + off; off += (size_t)Bn * Un * 128;
    float* kbuf   = wsf + off; off += (size_t)Bn * Un * 128;
    float* big0   = wsf + off; off += (size_t)Bn * Un * 2400;   // gx -> v -> ctxW
    float* big1   = wsf + off; off += (size_t)Bn * Un * 1800;   // gh
    size_t f32_bytes = off * sizeof(float);

    u16* wsb = (u16*)((char*)d_ws + ((f32_bytes + 15) & ~(size_t)15));
    size_t ob = 0;
    auto balloc = [&](size_t n) { u16* p = wsb + ob; ob += ((n + 23) & ~(size_t)7); return p; };
    // weights: K-tile-transposed padded  (U * padK * padN halves)
    u16* Wkb    = balloc((size_t)1  * padK(INn)  * padN(64));
    u16* Wvb    = balloc((size_t)1  * padK(INn)  * padN(IVS));
    u16* Wqb    = balloc((size_t)Un * padK(Hn)   * padN(IKS));
    u16* x2hb   = balloc((size_t)Un * padK(IVS)  * padN(1800));
    u16* h2hb   = balloc((size_t)Un * padK(Hn)   * padN(1800));
    u16* Wq_b   = balloc((size_t)Un * padK(Hn)   * padN(128));
    u16* Wk_b   = balloc((size_t)Un * padK(Hn)   * padN(128));
    u16* Wv_b   = balloc((size_t)Un * padK(Hn)   * padN(2400));
    u16* Woutb  = balloc((size_t)Un * padK(2400) * padN(Hn));
    // activations: row-major, K padded
    u16* xb     = balloc((size_t)Bn * INn);
    u16* hsb    = balloc((size_t)Bn * Un * Hp);
    u16* inputsb= balloc((size_t)Bn * Un * IVSp);
    u16* hsnewb = balloc((size_t)Bn * Un * Hp);
    u16* ctxb   = balloc((size_t)Bn * Un * 2400);
    size_t need = ((f32_bytes + 15) & ~(size_t)15) + ob * sizeof(u16);
    if (ws_size < need) return;   // scratch too small

    // 0) one-shot conversions
    cvtA(stream, x,  xb,  Bn,      INn, INn);
    cvtA(stream, hs, hsb, Bn * Un, Hn,  Hp);
    cvtW(stream, Wk,   Wkb,   1,  INn,  64);
    cvtW(stream, Wv,   Wvb,   1,  INn,  IVS);
    cvtW(stream, Wq,   Wqb,   Un, Hn,   IKS);
    cvtW(stream, x2h,  x2hb,  Un, IVS,  1800);
    cvtW(stream, h2h,  h2hb,  Un, Hn,   1800);
    cvtW(stream, Wq_,  Wq_b,  Un, Hn,   128);
    cvtW(stream, Wk_,  Wk_b,  Un, Hn,   128);
    cvtW(stream, Wv_,  Wv_b,  Un, Hn,   2400);
    cvtW(stream, Wout, Woutb, Un, 2400, Hn);

    // 1) key = x @ Wk + bk ; val = x @ Wv + bv
    gemm(stream, xb, 0, INn, Wkb, 0, padN(64),  bk, key, 0, 64,  Bn, 64,  padK(INn), 1);
    gemm(stream, xb, 0, INn, Wvb, 0, padN(IVS), bv, val, 0, IVS, Bn, IVS, padK(INn), 1);
    // 2) qry[u] = hs[:,u,:] @ Wq[u]
    gemm(stream, hsb, Hp, Un * Hp, Wqb, (ull)padK(Hn) * padN(IKS), padN(IKS), nullptr,
         qry, IKS, Un * IKS, Bn, IKS, padK(Hn), Un);
    // 3) scores -> top-k -> softmax -> inputs(bf16)
    scores_kernel<<<Bn / 8, 256, 0, stream>>>(qry, key, bk, val, bv, maskp, inputsb);
    // 4) gx = inputs @ x2h_w (big0) ; gh = hs @ h2h_w (big1)
    gemm(stream, inputsb, IVSp, Un * IVSp, x2hb, (ull)padK(IVS) * padN(1800), padN(1800), nullptr,
         big0, 1800, Un * 1800, Bn, 1800, padK(IVS), Un);
    gemm(stream, hsb, Hp, Un * Hp, h2hb, (ull)padK(Hn) * padN(1800), padN(1800), nullptr,
         big1, 1800, Un * 1800, Bn, 1800, padK(Hn), Un);
    // 5) GRU -> h_new (d_out fp32 + padded bf16 copy)
    gru_kernel<<<(Bn * Un * Hp + 255) / 256, 256, 0, stream>>>(big0, big1, hs, out, hsnewb);
    // 6) q/k/v from h_new
    gemm(stream, hsnewb, Hp, Un * Hp, Wq_b, (ull)padK(Hn) * padN(128), padN(128), nullptr,
         qbuf, 128, Un * 128, Bn, 128, padK(Hn), Un);
    gemm(stream, hsnewb, Hp, Un * Hp, Wk_b, (ull)padK(Hn) * padN(128), padN(128), nullptr,
         kbuf, 128, Un * 128, Bn, 128, padK(Hn), Un);
    gemm(stream, hsnewb, Hp, Un * Hp, Wv_b, (ull)padK(Hn) * padN(2400), padN(2400), nullptr,
         big0, 2400, Un * 2400, Bn, 2400, padK(Hn), Un);
    // 7) 4-head 6x6 attention -> ctx (bf16)
    attn_kernel<<<Bn / 8, 256, 0, stream>>>(qbuf, kbuf, big0, maskp, ctxb);
    // 8) ctxW = ctx @ Wout (reuse big0 as (B,U,600))
    gemm(stream, ctxb, 2400, Un * 2400, Woutb, (ull)padK(2400) * padN(Hn), padN(Hn), nullptr,
         big0, Hn, Un * Hn, Bn, Hn, padK(2400), Un);
    // 9) out = m*(ctxW + h_new) + (1-m)*h_old
    blend_kernel<<<(Bn * Un * Hn + 255) / 256, 256, 0, stream>>>(big0, hs, maskp, out);
}
